// AttentionBlock_8443905704547
// MI455X (gfx1250) — matrix operand order repaired and checked by reading, not compiled, not run on hardware
//
#include <hip/hip_runtime.h>

// ---------------------------------------------------------------------------
// AttentionBlock for MI455X (gfx1250, wave32, WMMA).
// S=4096, H=1024, NH=16, D=64. bf16 WMMA (16x16x32) GEMMs with fp32 accum,
// flash-attention fusion (probs never hit memory), fp32 softmax + LayerNorm.
// Workspace use: 64 MB (Xb, 4x Wt, Q/K/V, attn, pre-LN).
// ---------------------------------------------------------------------------

namespace {

constexpr int kS  = 4096;
constexpr int kH  = 1024;
constexpr int kNH = 16;
constexpr int kD  = 64;

typedef __attribute__((ext_vector_type(16))) __bf16 bf16x16;
typedef __attribute__((ext_vector_type(8)))  float  floatx8;

union BF16Frag { bf16x16 v; unsigned u[8]; unsigned short h16[16]; };
union F32Frag  { floatx8  v; float f[8]; };

__device__ __forceinline__ unsigned short f2bf(float f) {
  unsigned u = __float_as_uint(f);
  unsigned r = u + 0x7FFFu + ((u >> 16) & 1u);   // round-to-nearest-even
  return (unsigned short)(r >> 16);
}

__device__ __forceinline__ floatx8 wmma_bf16(bf16x16 a, bf16x16 b, floatx8 c) {
  // (neg_a, A, neg_b, B, c_mod, C, reuse_a, reuse_b)
  return __builtin_amdgcn_wmma_f32_16x16x32_bf16(false, a, false, b,
                                                 (short)0, c, false, false);
}

// A-matrix 16x32 bf16 fragment: lane = (half, m); VGPR v holds K pair:
//   v<4 : K = 2v + 8*half ; v>=4 : K = 16 + 2(v-4) + 8*half
// p points at element [row m][k-base]; K contiguous in memory.
__device__ __forceinline__ void load_afrag(BF16Frag& A, const unsigned short* p,
                                           int half) {
#pragma unroll
  for (int v = 0; v < 8; ++v) {
    int k = (v < 4) ? (2 * v + 8 * half) : (16 + 2 * (v - 4) + 8 * half);
    A.u[v] = *reinterpret_cast<const unsigned*>(p + k);
  }
}

// B-matrix 32x16 bf16 fragment: lane = (half, n); VGPR v holds K pair:
//   v<4 : K = 2v + 8*half ; v>=4 : K = 16 + 2(v-4) + 8*half
//   p points at element [col n][k-base] of K-major storage.
__device__ __forceinline__ void load_bfrag(BF16Frag& B, const unsigned short* p,
                                           int half) {
#pragma unroll
  for (int v = 0; v < 8; ++v) {
    int k = (v < 4) ? (2 * v + 8 * half) : (16 + 2 * (v - 4) + 8 * half);
    B.u[v] = *reinterpret_cast<const unsigned*>(p + k);
  }
}

__device__ __forceinline__ float red_max16(float v) {
#pragma unroll
  for (int m = 1; m < 16; m <<= 1) v = fmaxf(v, __shfl_xor(v, m, 32));
  return v;
}
__device__ __forceinline__ float red_sum16(float v) {
#pragma unroll
  for (int m = 1; m < 16; m <<= 1) v += __shfl_xor(v, m, 32);
  return v;
}

} // namespace

// ------------------------- conversion kernels ------------------------------

__global__ __launch_bounds__(256) void k_conv_x(const float* __restrict__ x,
                                                unsigned short* __restrict__ xb,
                                                int n) {
  int i = blockIdx.x * 256 + threadIdx.x;
  if (i < n) xb[i] = f2bf(x[i]);
}

// Wt[j*kH + k] = bf16(W[k*kH + j])  (K-major for B fragments)
__global__ __launch_bounds__(256) void k_conv_wt(const float* __restrict__ w,
                                                 unsigned short* __restrict__ wt) {
  int i = blockIdx.x * 256 + threadIdx.x;   // over kH*kH
  int j = i & (kH - 1);
  int k = i >> 10;
  wt[j * kH + k] = f2bf(w[k * kH + j]);
}

// ------------------------- QKV projection GEMM -----------------------------
// C = bf16(X) @ W + b, written bf16 in head-major [h][s][d] layout.
// 4 waves / block; block tile 64x64; wave strip 16 rows x 64 cols.
__global__ __launch_bounds__(128) void k_gemm_qkv(
    const unsigned short* __restrict__ Xb, const unsigned short* __restrict__ Wt,
    const float* __restrict__ bias, unsigned short* __restrict__ out_hsd) {
  int lane = threadIdx.x & 31, wave = threadIdx.x >> 5;
  int half = lane >> 4, ln16 = lane & 15;
  int row0 = blockIdx.x * 64 + wave * 16;
  int col0 = blockIdx.y * 64;

  F32Frag acc[4];
#pragma unroll
  for (int s = 0; s < 4; ++s)
#pragma unroll
    for (int r = 0; r < 8; ++r) acc[s].f[r] = 0.0f;

  const unsigned short* arow = Xb + (size_t)(row0 + ln16) * kH;
  for (int kk = 0; kk < kH; kk += 32) {
    BF16Frag A;
    load_afrag(A, arow + kk, half);
#pragma unroll
    for (int s = 0; s < 4; ++s) {
      BF16Frag B;
      load_bfrag(B, Wt + (size_t)(col0 + 16 * s + ln16) * kH + kk, half);
      acc[s].v = wmma_bf16(A.v, B.v, acc[s].v);
    }
  }
#pragma unroll
  for (int s = 0; s < 4; ++s)
#pragma unroll
    for (int r = 0; r < 8; ++r) {
      int row = row0 + r + 8 * half;
      int col = col0 + 16 * s + ln16;
      float val = acc[s].f[r] + bias[col];
      int h = col >> 6, d = col & (kD - 1);
      out_hsd[((size_t)h * kS + row) * kD + d] = f2bf(val);
    }
}

// ------------------------- flash attention ---------------------------------
// grid (kS/64, kNH), 128 threads (4 waves). Each wave owns 16 query rows.
__global__ __launch_bounds__(128) void k_flash_attn(
    const unsigned short* __restrict__ Qb, const unsigned short* __restrict__ Kb,
    const unsigned short* __restrict__ Vb, const int* __restrict__ mask,
    unsigned short* __restrict__ attn) {
  __shared__ unsigned short ldsVt[64 * 66];       // V tile, transposed [d][kk]
  __shared__ unsigned short ldsP[4 * 16 * 66];    // per-wave P strips

  int lane = threadIdx.x & 31, wave = threadIdx.x >> 5;
  int half = lane >> 4, ln16 = lane & 15;
  int h  = blockIdx.y;
  int q0 = blockIdx.x * 64;

  const unsigned short* Qh = Qb + (size_t)h * kS * kD;
  const unsigned short* Kh = Kb + (size_t)h * kS * kD;
  const unsigned short* Vh = Vb + (size_t)h * kS * kD;

  // Query A-fragments (K = 64 -> two 32-wide k-steps), resident all loop.
  int qrow = q0 + wave * 16 + ln16;
  BF16Frag QA0, QA1;
  load_afrag(QA0, Qh + (size_t)qrow * kD + 0, half);
  load_afrag(QA1, Qh + (size_t)qrow * kD + 32, half);

  const float scale = 0.03125f;                    // 1/sqrt(H)
  float mq[8], mrow[8], lrow[8];
  F32Frag O[4];
#pragma unroll
  for (int r = 0; r < 8; ++r) {
    mq[r]   = (float)mask[q0 + wave * 16 + r + 8 * half];
    mrow[r] = -3.0e38f;
    lrow[r] = 0.0f;
  }
#pragma unroll
  for (int s = 0; s < 4; ++s)
#pragma unroll
    for (int r = 0; r < 8; ++r) O[s].f[r] = 0.0f;

  unsigned short* Pw = ldsP + wave * 16 * 66;

  for (int k0 = 0; k0 < kS; k0 += 64) {
    __syncthreads();                               // protect prior Vt reads
    for (int t = threadIdx.x; t < 64 * 64; t += 128) {
      int d = t & 63, kr = t >> 6;
      ldsVt[d * 66 + kr] = Vh[(size_t)(k0 + kr) * kD + d];
    }
    __syncthreads();

    // ---- scores: S = Q @ K^T -------------------------------------------
    F32Frag SC[4];
#pragma unroll
    for (int s = 0; s < 4; ++s)
#pragma unroll
      for (int r = 0; r < 8; ++r) SC[s].f[r] = 0.0f;

#pragma unroll
    for (int s = 0; s < 4; ++s) {
      const unsigned short* kcol = Kh + (size_t)(k0 + 16 * s + ln16) * kD;
      BF16Frag B0, B1;
      load_bfrag(B0, kcol + 0, half);
      load_bfrag(B1, kcol + 32, half);
      SC[s].v = wmma_bf16(QA0.v, B0.v, SC[s].v);
      SC[s].v = wmma_bf16(QA1.v, B1.v, SC[s].v);
    }

    // ---- multiplicative mask + scale (exactly as reference) -------------
    float mk[4];
#pragma unroll
    for (int s = 0; s < 4; ++s)
      mk[s] = (float)mask[k0 + 16 * s + ln16];
#pragma unroll
    for (int s = 0; s < 4; ++s)
#pragma unroll
      for (int r = 0; r < 8; ++r) SC[s].f[r] *= scale * mq[r] * mk[s];

    // ---- online softmax (fp32) ------------------------------------------
    float alpha[8];
#pragma unroll
    for (int r = 0; r < 8; ++r) {
      float t = SC[0].f[r];
      t = fmaxf(t, SC[1].f[r]);
      t = fmaxf(t, SC[2].f[r]);
      t = fmaxf(t, SC[3].f[r]);
      t = red_max16(t);
      float mnew = fmaxf(mrow[r], t);
      alpha[r] = __expf(mrow[r] - mnew);
      mrow[r]  = mnew;
    }
    float rsum[8];
#pragma unroll
    for (int r = 0; r < 8; ++r) rsum[r] = 0.0f;
#pragma unroll
    for (int s = 0; s < 4; ++s)
#pragma unroll
      for (int r = 0; r < 8; ++r) {
        float p = __expf(SC[s].f[r] - mrow[r]);
        SC[s].f[r] = p;
        rsum[r] += p;
      }
#pragma unroll
    for (int r = 0; r < 8; ++r) {
      lrow[r] = lrow[r] * alpha[r] + red_sum16(rsum[r]);
    }
#pragma unroll
    for (int s = 0; s < 4; ++s)
#pragma unroll
      for (int r = 0; r < 8; ++r) O[s].f[r] *= alpha[r];

    // ---- P: C-layout -> A-layout via this wave's private LDS strip ------
#pragma unroll
    for (int s = 0; s < 4; ++s)
#pragma unroll
      for (int r = 0; r < 8; ++r)
        Pw[(r + 8 * half) * 66 + 16 * s + ln16] = f2bf(SC[s].f[r]);

    // ---- O += P @ V ------------------------------------------------------
#pragma unroll
    for (int ks = 0; ks < 2; ++ks) {
      BF16Frag PA;
      load_afrag(PA, Pw + ln16 * 66 + 32 * ks, half);
#pragma unroll
      for (int s = 0; s < 4; ++s) {
        BF16Frag VB;
        load_bfrag(VB, ldsVt + (16 * s + ln16) * 66 + 32 * ks, half);
        O[s].v = wmma_bf16(PA.v, VB.v, O[s].v);
      }
    }
  }

  // epilogue: attn[s][h*D + d] = O / l  (bf16 for the output projection)
#pragma unroll
  for (int s = 0; s < 4; ++s)
#pragma unroll
    for (int r = 0; r < 8; ++r) {
      int row = q0 + wave * 16 + r + 8 * half;
      int col = h * kD + 16 * s + ln16;
      attn[(size_t)row * kH + col] = f2bf(O[s].f[r] / lrow[r]);
    }
}

// ------------------ output projection + bias + residual --------------------
__global__ __launch_bounds__(128) void k_gemm_out(
    const unsigned short* __restrict__ Ab, const unsigned short* __restrict__ Wt,
    const float* __restrict__ bias, const float* __restrict__ X,
    float* __restrict__ preln) {
  int lane = threadIdx.x & 31, wave = threadIdx.x >> 5;
  int half = lane >> 4, ln16 = lane & 15;
  int row0 = blockIdx.x * 64 + wave * 16;
  int col0 = blockIdx.y * 64;

  F32Frag acc[4];
#pragma unroll
  for (int s = 0; s < 4; ++s)
#pragma unroll
    for (int r = 0; r < 8; ++r) acc[s].f[r] = 0.0f;

  const unsigned short* arow = Ab + (size_t)(row0 + ln16) * kH;
  for (int kk = 0; kk < kH; kk += 32) {
    BF16Frag A;
    load_afrag(A, arow + kk, half);
#pragma unroll
    for (int s = 0; s < 4; ++s) {
      BF16Frag B;
      load_bfrag(B, Wt + (size_t)(col0 + 16 * s + ln16) * kH + kk, half);
      acc[s].v = wmma_bf16(A.v, B.v, acc[s].v);
    }
  }
#pragma unroll
  for (int s = 0; s < 4; ++s)
#pragma unroll
    for (int r = 0; r < 8; ++r) {
      int row = row0 + r + 8 * half;
      int col = col0 + 16 * s + ln16;
      size_t idx = (size_t)row * kH + col;
      preln[idx] = acc[s].f[r] + bias[col] + X[idx];
    }
}

// ---------------------------- LayerNorm (fp32) -----------------------------
__global__ __launch_bounds__(256) void k_layernorm(
    const float* __restrict__ preln, const float* __restrict__ lnw,
    const float* __restrict__ lnb, float* __restrict__ out) {
  __shared__ float s1[256], s2[256];
  int row = blockIdx.x, tid = threadIdx.x;
  const float* x = preln + (size_t)row * kH;
  float sum = 0.0f, sum2 = 0.0f;
  for (int j = tid; j < kH; j += 256) {
    float v = x[j];
    sum += v;
    sum2 += v * v;
  }
  s1[tid] = sum; s2[tid] = sum2;
  __syncthreads();
  for (int off = 128; off > 0; off >>= 1) {
    if (tid < off) { s1[tid] += s1[tid + off]; s2[tid] += s2[tid + off]; }
    __syncthreads();
  }
  float mean = s1[0] * (1.0f / kH);
  float var  = s2[0] * (1.0f / kH) - mean * mean;
  float inv  = rsqrtf(var + 1e-5f);
  float* o = out + (size_t)row * kH;
  for (int j = tid; j < kH; j += 256)
    o[j] = (x[j] - mean) * inv * lnw[j] + lnb[j];
}

// ------------------------------- launcher ----------------------------------
extern "C" void kernel_launch(void* const* d_in, const int* in_sizes, int n_in,
                              void* d_out, int out_size, void* d_ws,
                              size_t ws_size, hipStream_t stream) {
  (void)in_sizes; (void)n_in; (void)out_size; (void)ws_size;
  const float* X    = (const float*)d_in[0];
  const int*   mask = (const int*)d_in[1];
  const float* Wq = (const float*)d_in[2];  const float* bq = (const float*)d_in[3];
  const float* Wk = (const float*)d_in[4];  const float* bk = (const float*)d_in[5];
  const float* Wv = (const float*)d_in[6];  const float* bv = (const float*)d_in[7];
  const float* Wo = (const float*)d_in[8];  const float* bo = (const float*)d_in[9];
  const float* lnw = (const float*)d_in[10];
  const float* lnb = (const float*)d_in[11];
  float* out = (float*)d_out;

  char* ws = (char*)d_ws;
  unsigned short* Xb   = (unsigned short*)(ws + 0);          //  8 MB
  unsigned short* Wtq  = (unsigned short*)(ws + 8388608);    //  2 MB
  unsigned short* Wtk  = (unsigned short*)(ws + 10485760);   //  2 MB
  unsigned short* Wtv  = (unsigned short*)(ws + 12582912);   //  2 MB
  unsigned short* Wto  = (unsigned short*)(ws + 14680064);   //  2 MB
  unsigned short* Qb   = (unsigned short*)(ws + 16777216);   //  8 MB
  unsigned short* Kb   = (unsigned short*)(ws + 25165824);   //  8 MB
  unsigned short* Vb   = (unsigned short*)(ws + 33554432);   //  8 MB
  unsigned short* Attn = (unsigned short*)(ws + 41943040);   //  8 MB
  float*          Pre  = (float*)(ws + 50331648);            // 16 MB

  dim3 b256(256), b128(128);
  k_conv_x<<<(kS * kH + 255) / 256, b256, 0, stream>>>(X, Xb, kS * kH);
  k_conv_wt<<<(kH * kH) / 256, b256, 0, stream>>>(Wq, Wtq);
  k_conv_wt<<<(kH * kH) / 256, b256, 0, stream>>>(Wk, Wtk);
  k_conv_wt<<<(kH * kH) / 256, b256, 0, stream>>>(Wv, Wtv);
  k_conv_wt<<<(kH * kH) / 256, b256, 0, stream>>>(Wo, Wto);

  dim3 gGemm(kS / 64, kH / 64);
  k_gemm_qkv<<<gGemm, b128, 0, stream>>>(Xb, Wtq, bq, Qb);
  k_gemm_qkv<<<gGemm, b128, 0, stream>>>(Xb, Wtk, bk, Kb);
  k_gemm_qkv<<<gGemm, b128, 0, stream>>>(Xb, Wtv, bv, Vb);

  dim3 gAttn(kS / 64, kNH);
  k_flash_attn<<<gAttn, b128, 0, stream>>>(Qb, Kb, Vb, mask, Attn);

  k_gemm_out<<<gGemm, b128, 0, stream>>>(Attn, Wto, bo, X, Pre);
  k_layernorm<<<kS, b256, 0, stream>>>(Pre, lnw, lnb, out);
}
